// NodeAttentionLayer_22411139350655
// MI455X (gfx1250) — matrix operand order repaired and checked by reading, not compiled, not run on hardware
//
#include <hip/hip_runtime.h>
#include <hip/hip_bf16.h>
#include <math.h>

// ---------------------------------------------------------------------------
// NodeAttentionLayer for MI455X (gfx1250, wave32, WMMA).
//
// Roofline: main GEMM (400000x128 @ 128x128) reads 205 MB of h_j -> ~8.8us at
// 23.3 TB/s. fp32 WMMA (16x16x4) would be compute-bound; converting h_j tiles
// to bf16 in-kernel and using v_wmma_f32_16x16x32_bf16 makes the pass memory-
// bound. Online (flash-style) softmax keeps it to ONE pass over h_j: each
// block emits {local max, local expsum, 128-float weighted partial}, tiny
// kernels combine them. All reductions fixed-order => deterministic.
// ---------------------------------------------------------------------------

#define N_J   400000
#define DIM   128            // O_DIM == T_DIM == OUT_F
#define BM    128            // rows of h_j per block
#define NB    (N_J / BM)     // 3125 (exact)
#define LDSTR 136            // padded LDS row stride in bf16 elems (272 B)

typedef __bf16 v16bf __attribute__((ext_vector_type(16)));
typedef __bf16 v8bf  __attribute__((ext_vector_type(8)));
typedef __bf16 v4bf  __attribute__((ext_vector_type(4)));
typedef float  v8f   __attribute__((ext_vector_type(8)));

// Workspace layout (floats unless noted):
//  [0]=sl  [1]=gmax  [2]=gsum ; byte 64..64+32768: W_j_w in bf16 (compact)
//  float idx WS_F0 = (64+32768)/4: maxb[NB], sumb[NB], sc[NB], ph[128*NB]
#define WS_F0    8208
#define WS_MAXB  WS_F0
#define WS_SUMB  (WS_F0 + NB)
#define WS_SC    (WS_F0 + 2*NB)
#define WS_PH    (WS_F0 + 3*NB)

// ---------------------------------------------------------------------------
// Kernel 1: z_i = h_i@W_i_w^T + b ; sl = dot(w_l, z_i) ; W_j_w -> bf16
// ---------------------------------------------------------------------------
__global__ __launch_bounds__(256) void k_prep(
    const float* __restrict__ h_i, const float* __restrict__ W_i_w,
    const float* __restrict__ W_i_b, const float* __restrict__ W_j_w,
    const float* __restrict__ W_ij, float* __restrict__ ws_f,
    __bf16* __restrict__ Wbf) {
  __shared__ float sh[DIM];
  const int t = threadIdx.x;
  if (t < DIM) {
    float z = W_i_b[t];
    const float* wrow = W_i_w + t * DIM;
    for (int k = 0; k < DIM; ++k) z = fmaf(h_i[k], wrow[k], z);
    sh[t] = z * W_ij[t];            // w_l[t] * z_i[t]
  }
  __syncthreads();
  if (t == 0) {
    float s = 0.f;
    for (int i = 0; i < DIM; ++i) s += sh[i];
    ws_f[0] = s;                    // sl (fixed-order, deterministic)
  }
  for (int i = t; i < DIM * DIM; i += 256) Wbf[i] = (__bf16)W_j_w[i];
}

// ---------------------------------------------------------------------------
// Kernel 2: main fused GEMM + scores + online softmax partials.
// 256 threads = 8 waves; wave w owns rows [b*128 + w*16, +16).
// ---------------------------------------------------------------------------
__global__ __launch_bounds__(256) void k_main(
    const float* __restrict__ h_j, const __bf16* __restrict__ Wbf,
    const float* __restrict__ W_j_b, const float* __restrict__ W_ij,
    const float* __restrict__ ws_f, float* __restrict__ maxb_o,
    float* __restrict__ sumb_o, float* __restrict__ ph_o) {
  __shared__ __bf16 Ash[BM  * LDSTR];   // h_j tile, bf16
  __shared__ __bf16 Wsh[DIM * LDSTR];   // W_j_w,  bf16
  __shared__ float  e_sh[BM];
  __shared__ float  bias_sh[DIM];
  __shared__ float  wr_sh[DIM];
  __shared__ float  ph_wave[8][DIM];
  __shared__ float  red[2];

  const int b    = blockIdx.x;
  const int tid  = threadIdx.x;
  const int wave = tid >> 5, lane = tid & 31;
  const int half = lane >> 4, l16 = lane & 15;

  // ---- stage W (bf16, 16B vectors) ----
  #pragma unroll
  for (int it = 0; it < 8; ++it) {
    int j = tid + 256 * it;                 // v8bf index (16 per row)
    int row = j >> 4, col = (j & 15) * 8;
    *(v8bf*)&Wsh[row * LDSTR + col] = ((const v8bf*)Wbf)[j];
  }
  // ---- stage h_j tile: f32 -> bf16 ----
  const float4* hj4 = (const float4*)(h_j + (size_t)b * (BM * DIM));
  #pragma unroll
  for (int it = 0; it < 16; ++it) {
    int j = tid + 256 * it;                 // float4 index (32 per row)
    float4 v = hj4[j];
    int row = j >> 5, col = (j & 31) * 4;
    v4bf o;
    o[0] = (__bf16)v.x; o[1] = (__bf16)v.y; o[2] = (__bf16)v.z; o[3] = (__bf16)v.w;
    *(v4bf*)&Ash[row * LDSTR + col] = o;
  }
  if (tid < DIM) { bias_sh[tid] = W_j_b[tid]; wr_sh[tid] = W_ij[DIM + tid]; }
  __syncthreads();

  // ---- WMMA GEMM: 8 f-tiles x 4 k-steps per wave ----
  v8f acc[8];
  #pragma unroll
  for (int ft = 0; ft < 8; ++ft)
    #pragma unroll
    for (int v = 0; v < 8; ++v) acc[ft][v] = 0.f;

  const int arow = wave * 16 + l16;         // this lane's A-matrix row (M)
  #pragma unroll
  for (int kk = 0; kk < 4; ++kk) {
    const int k0 = kk * 32;
    // A 16x32 bf16: lanes 0-15 K={k0..k0+7, k0+16..23}; lanes 16-31 shifted +8
    const __bf16* ap = &Ash[arow * LDSTR + k0 + half * 8];
    v8bf a0 = *(const v8bf*)ap;
    v8bf a1 = *(const v8bf*)(ap + 16);
    v16bf af;
    #pragma unroll
    for (int i = 0; i < 8; ++i) { af[i] = a0[i]; af[i + 8] = a1[i]; }
    #pragma unroll
    for (int ft = 0; ft < 8; ++ft) {
      // B 32x16 bf16: lane = N = f-col; lanes 0-15 K={k0..k0+7, k0+16..23}; lanes 16-31 shifted +8
      const __bf16* bp = &Wsh[(ft * 16 + l16) * LDSTR + k0 + half * 8];
      v8bf b0 = *(const v8bf*)bp;
      v8bf b1 = *(const v8bf*)(bp + 16);
      v16bf bfr;
      #pragma unroll
      for (int i = 0; i < 8; ++i) { bfr[i] = b0[i]; bfr[i + 8] = b1[i]; }
      acc[ft] = __builtin_amdgcn_wmma_f32_16x16x32_bf16(
          false, af, false, bfr, (short)0, acc[ft], false, false);
    }
  }

  // ---- bias + score partials: lane holds Z[m, f] for m=half*8+v, f=ft*16+l16
  const float sl = ws_f[0];
  float p[8] = {0, 0, 0, 0, 0, 0, 0, 0};
  #pragma unroll
  for (int ft = 0; ft < 8; ++ft) {
    const int f = ft * 16 + l16;
    const float bv = bias_sh[f], wv = wr_sh[f];
    #pragma unroll
    for (int v = 0; v < 8; ++v) {
      float z = acc[ft][v] + bv;
      acc[ft][v] = z;
      p[v] = fmaf(z, wv, p[v]);
    }
  }
  // reduce scores across the 16 lanes of each half (fixed order)
  #pragma unroll
  for (int m = 1; m < 16; m <<= 1)
    #pragma unroll
    for (int v = 0; v < 8; ++v) p[v] += __shfl_xor(p[v], m, 32);
  if (l16 == 0) {
    #pragma unroll
    for (int v = 0; v < 8; ++v) {
      float s = p[v] + sl;
      e_sh[wave * 16 + half * 8 + v] = (s > 0.f) ? s : 0.1f * s;  // leaky relu
    }
  }
  __syncthreads();

  // ---- block max / expsum (wave 0, shuffle tree) ----
  if (wave == 0) {
    float m = -INFINITY;
    #pragma unroll
    for (int i = 0; i < 4; ++i) m = fmaxf(m, e_sh[lane + 32 * i]);
    #pragma unroll
    for (int x = 1; x < 32; x <<= 1) m = fmaxf(m, __shfl_xor(m, x, 32));
    float s = 0.f;
    #pragma unroll
    for (int i = 0; i < 4; ++i) s += expf(e_sh[lane + 32 * i] - m);
    #pragma unroll
    for (int x = 1; x < 32; x <<= 1) s += __shfl_xor(s, x, 32);
    if (lane == 0) { red[0] = m; red[1] = s; }
  }
  __syncthreads();

  // ---- weighted partial vector: ph[f] = sum_m exp(e_m - maxb) * Z[m,f] ----
  const float mb = red[0];
  float ev[8];
  #pragma unroll
  for (int v = 0; v < 8; ++v)
    ev[v] = expf(e_sh[wave * 16 + half * 8 + v] - mb);
  #pragma unroll
  for (int ft = 0; ft < 8; ++ft) {
    float ph = 0.f;
    #pragma unroll
    for (int v = 0; v < 8; ++v) ph = fmaf(ev[v], acc[ft][v], ph);
    ph += __shfl_xor(ph, 16, 32);           // fold the two row-halves
    if (half == 0) ph_wave[wave][ft * 16 + l16] = ph;
  }
  __syncthreads();
  if (tid < DIM) {
    float h = 0.f;
    #pragma unroll
    for (int w = 0; w < 8; ++w) h += ph_wave[w][tid];   // fixed order
    ph_o[(size_t)tid * NB + b] = h;         // transposed -> coalesced reduce
    if (tid == 0) { maxb_o[b] = red[0]; sumb_o[b] = red[1]; }
  }
}

// ---------------------------------------------------------------------------
// Kernel 3: global softmax scalars: gmax, sc[b]=exp(maxb-gmax), gsum.
// ---------------------------------------------------------------------------
__global__ __launch_bounds__(256) void k_soft(float* __restrict__ ws_f) {
  __shared__ float sh[256];
  const int t = threadIdx.x;
  float* maxb = ws_f + WS_MAXB;
  float* sumb = ws_f + WS_SUMB;
  float* sc   = ws_f + WS_SC;
  float m = -INFINITY;
  for (int b = t; b < NB; b += 256) m = fmaxf(m, maxb[b]);
  sh[t] = m;
  __syncthreads();
  float gm = -INFINITY;
  for (int i = 0; i < 256; ++i) gm = fmaxf(gm, sh[i]);
  __syncthreads();
  float s = 0.f;
  for (int b = t; b < NB; b += 256) {
    float e = expf(maxb[b] - gm);
    sc[b] = e;
    s += sumb[b] * e;
  }
  sh[t] = s;
  __syncthreads();
  if (t == 0) {
    float gs = 0.f;
    for (int i = 0; i < 256; ++i) gs += sh[i];
    ws_f[1] = gm;
    ws_f[2] = gs;
  }
}

// ---------------------------------------------------------------------------
// Kernel 4: per-feature combine: h[f] = sum_b ph[f][b]*sc[b] / gsum ; ELU out.
// One block per feature; coalesced reads of the transposed partials.
// ---------------------------------------------------------------------------
__global__ __launch_bounds__(256) void k_final(const float* __restrict__ ws_f,
                                               float* __restrict__ out) {
  __shared__ float sh[256];
  const int f = blockIdx.x, t = threadIdx.x;
  const float* ph = ws_f + WS_PH + (size_t)f * NB;
  const float* sc = ws_f + WS_SC;
  float a = 0.f;
  for (int b = t; b < NB; b += 256) a = fmaf(ph[b], sc[b], a);
  sh[t] = a;
  __syncthreads();
  if (t == 0) {
    float h = 0.f;
    for (int i = 0; i < 256; ++i) h += sh[i];
    h /= ws_f[2];
    out[f] = (h > 0.f) ? h : expm1f(h);
  }
}

// ---------------------------------------------------------------------------
extern "C" void kernel_launch(void* const* d_in, const int* in_sizes, int n_in,
                              void* d_out, int out_size, void* d_ws,
                              size_t ws_size, hipStream_t stream) {
  (void)in_sizes; (void)n_in; (void)out_size; (void)ws_size;
  const float* h_i   = (const float*)d_in[0];
  const float* h_j   = (const float*)d_in[1];
  const float* W_i_w = (const float*)d_in[2];
  const float* W_i_b = (const float*)d_in[3];
  const float* W_j_w = (const float*)d_in[4];
  const float* W_j_b = (const float*)d_in[5];
  const float* W_ij  = (const float*)d_in[6];
  float*   ws_f = (float*)d_ws;
  __bf16*  Wbf  = (__bf16*)((char*)d_ws + 64);
  float*   out  = (float*)d_out;

  hipLaunchKernelGGL(k_prep, dim3(1), dim3(256), 0, stream,
                     h_i, W_i_w, W_i_b, W_j_w, W_ij, ws_f, Wbf);
  hipLaunchKernelGGL(k_main, dim3(NB), dim3(256), 0, stream,
                     h_j, Wbf, W_j_b, W_ij, ws_f,
                     ws_f + WS_MAXB, ws_f + WS_SUMB, ws_f + WS_PH);
  hipLaunchKernelGGL(k_soft, dim3(1), dim3(256), 0, stream, ws_f);
  hipLaunchKernelGGL(k_final, dim3(DIM), dim3(256), 0, stream, ws_f, out);
}
